// BiLSTM_79731772883680
// MI455X (gfx1250) — matrix operand order repaired and checked by reading, not compiled, not run on hardware
//
#include <hip/hip_runtime.h>
#include <hip/hip_bf16.h>
#include <cstdint>

#define T_LEN   256
#define BATCH   64
#define EMB     512
#define HDIM    256
#define G4      1024   // 4*HD
#define K_TAGS  7
#define START_TAG 5
#define STOP_TAG  6
#define NEGINF  -10000.0f

typedef __attribute__((ext_vector_type(16))) __bf16 v16bf;
typedef __attribute__((ext_vector_type(8)))  float  v8f;

union FragAB { v16bf v; unsigned int u[8]; };
union FragC  { v8f   v; float        f[8]; };

__device__ __forceinline__ unsigned int pack2bf(float a, float b) {
    unsigned int ua = __float_as_uint(a), ub = __float_as_uint(b);
    ua = (ua + 0x7FFFu + ((ua >> 16) & 1u)) >> 16;   // RNE f32 -> bf16
    ub = (ub + 0x7FFFu + ((ub >> 16) & 1u)) >> 16;
    return (ua & 0xFFFFu) | (ub << 16);
}
__device__ __forceinline__ float sigf(float x) { return 1.0f / (1.0f + __expf(-x)); }

// ---- embedding gather, emitting packed-bf16 rows: xbf[row][256 dwords] ----
__global__ void k_embed(const int* __restrict__ sent, const float* __restrict__ emb,
                        unsigned int* __restrict__ xbf) {
    const int row = blockIdx.x;                 // t*B + b
    const int tok = sent[row];
    const float4 f = ((const float4*)(emb + (size_t)tok * EMB))[threadIdx.x];
    unsigned int* dst = xbf + (size_t)row * 256 + threadIdx.x * 2;
    dst[0] = pack2bf(f.x, f.y);
    dst[1] = pack2bf(f.z, f.w);
}

// ---- pack W_hh (f32) -> bf16 WMMA B-fragment layout: [ld][64nt][8kt][32][8] ----
__global__ void k_prep_whh(const float* __restrict__ w_hh, unsigned int* __restrict__ whhp) {
    int id = blockIdx.x * blockDim.x + threadIdx.x;
    if (id >= 4 * 131072) return;
    int v = id & 7, lane = (id >> 3) & 31, kt = (id >> 8) & 7, nt = (id >> 11) & 63, ld = id >> 17;
    int half = lane >> 4;
    int n = nt * 16 + (lane & 15);
    int k = kt * 32 + 8 * half + 2 * (v & 3) + 16 * (v >> 2);
    const float* w = w_hh + (size_t)ld * G4 * HDIM + (size_t)n * HDIM + k;
    whhp[id] = pack2bf(w[0], w[1]);
}

// ---- pack W_ih (f32) -> bf16 WMMA B-fragment layout: [ld][64nt][16kt][32][8] ----
__global__ void k_prep_wih(const float* __restrict__ w_ih, unsigned int* __restrict__ wihp) {
    int id = blockIdx.x * blockDim.x + threadIdx.x;
    if (id >= 4 * 262144) return;
    int v = id & 7, lane = (id >> 3) & 31, kt = (id >> 8) & 15, nt = (id >> 12) & 63, ld = id >> 18;
    int half = lane >> 4;
    int n = nt * 16 + (lane & 15);
    int k = kt * 32 + 16 * half + 2 * v;
    const float* w = w_ih + (size_t)ld * G4 * EMB + (size_t)n * EMB + k;
    wihp[id] = pack2bf(w[0], w[1]);
}

// ------- xg = x @ W_ih^T + b_ih + b_hh -------
// A tiles DMA'd into LDS via GLOBAL_LOAD_ASYNC_TO_LDS_B128 (double buffered,
// ASYNCcnt-tracked); B fragments read straight from prepacked global (L2-resident).
// grid: (M/16, N/64, 2 dirs), block 128 = 4 waves, one 16x16 N-tile each.
__global__ void k_gemm_xw(const unsigned int* __restrict__ xbf,    // [16384][256] dwords
                          const unsigned int* __restrict__ wihp_l, // [2][64][16][32][8]
                          const float* __restrict__ bih_l, const float* __restrict__ bhh_l,
                          float* __restrict__ xg0, float* __restrict__ xg1) {
    const int dir = blockIdx.z;
    const unsigned int* wihp = wihp_l + (size_t)dir * 262144;
    const float* bi = bih_l + dir * G4;
    const float* bh = bhh_l + dir * G4;
    float* xg = dir ? xg1 : xg0;

    const int m0 = blockIdx.x * 16;
    const int tid = threadIdx.x;
    const int lane = tid & 31, wv = tid >> 5;
    const int half = lane >> 4, l16 = lane & 15;
    const int ntg = blockIdx.y * 4 + wv;

    __shared__ unsigned int As[2][256];         // two 16x32 bf16 tiles (1KB each)
    const unsigned lbase = (unsigned)(uintptr_t)&As[0][0];

    // async-DMA one A tile (1KB) into LDS: waves 0&1, 16B per lane
    auto issueA = [&](int kt, int bufi) {
        if (tid < 64) {
            unsigned lds = lbase + (unsigned)bufi * 1024u + (unsigned)tid * 16u;
            unsigned long long ga = (unsigned long long)(uintptr_t)
                ((const char*)xbf + (size_t)(m0 + (tid >> 2)) * 1024 + kt * 64 + (tid & 3) * 16);
            asm volatile("global_load_async_to_lds_b128 %0, %1, off"
                         :: "v"(lds), "v"(ga) : "memory");
        }
    };

    FragC acc;
#pragma unroll
    for (int i = 0; i < 8; ++i) acc.f[i] = 0.0f;

    issueA(0, 0);
    for (int kt = 0; kt < 16; ++kt) {
        if (kt + 1 < 16) issueA(kt + 1, (kt + 1) & 1);
        if (tid < 64) {
            if (kt + 1 < 16) asm volatile("s_wait_asynccnt 0x1" ::: "memory");
            else             asm volatile("s_wait_asynccnt 0x0" ::: "memory");
        }
        __syncthreads();
        const unsigned int* As32 = As[kt & 1];
        const unsigned int* bp = wihp + (((size_t)ntg * 16 + kt) * 32 + lane) * 8;
        FragAB a, bf;
#pragma unroll
        for (int v = 0; v < 8; ++v) {
            int dwA = v + 8 * half;                      // contiguous k = 16*half + 2v
            a.u[v]  = As32[l16 * 16 + dwA];
            bf.u[v] = bp[v];
        }
        acc.v = __builtin_amdgcn_wmma_f32_16x16x32_bf16(false, a.v, false, bf.v,
                                                        (short)0, acc.v, false, false);
        __syncthreads();
    }
    const int col = ntg * 16 + l16;
    const float bias = bi[col] + bh[col];
#pragma unroll
    for (int v = 0; v < 8; ++v) {
        int m = m0 + v + 8 * half;                       // f32 C/D layout
        xg[(size_t)m * G4 + col] = acc.f[v] + bias;
    }
}

// --------- persistent recurrent scan: one 512-thread block per direction ---------
// LDS: gbuf f32[64][1024] + hA bf16 A-fragments [4mt][8kt][32][8]  (288 KB)
__global__ void k_lstm_scan(const float* __restrict__ xg0, const float* __restrict__ xg1,
                            const unsigned int* __restrict__ whhp_l, // [2][64][8][32][8]
                            const float* __restrict__ h0_l, const float* __restrict__ c0_l,
                            float* __restrict__ xout,                // [T][B][512] f32
                            unsigned int* __restrict__ xoutbf) {     // [T*B][256] packed bf16
    const int dir = blockIdx.x;
    const bool rev = (dir == 1);
    const float* xg = dir ? xg1 : xg0;
    const unsigned int* whhp = whhp_l + (size_t)dir * 131072;
    const float* h0d = h0_l + (size_t)dir * BATCH * HDIM;
    const float* c0d = c0_l + (size_t)dir * BATCH * HDIM;

    extern __shared__ unsigned char smem[];
    float* gbuf = (float*)smem;
    unsigned int* hA = (unsigned int*)(smem + (size_t)BATCH * G4 * 4);

    const int tid = threadIdx.x;
    const int lane = tid & 31, wave = tid >> 5;
    const int half = lane >> 4, l16 = lane & 15;
    const int mt = wave & 3, ntb = wave >> 2;
    const int b = tid >> 3, kto = tid & 7, hd0 = kto * 32;

    float creg[32];
#pragma unroll
    for (int j = 0; j < 32; ++j) creg[j] = c0d[b * HDIM + hd0 + j];
#pragma unroll
    for (int j = 0; j < 32; j += 2) {
        int la = (b & 15) + 16 * ((j >> 3) & 1);
        int v  = ((j & 7) >> 1) + 4 * (j >> 4);
        hA[(((b >> 4) * 8 + kto) * 32 + la) * 8 + v] =
            pack2bf(h0d[b * HDIM + hd0 + j], h0d[b * HDIM + hd0 + j + 1]);
    }
    __syncthreads();

    for (int step = 0; step < T_LEN; ++step) {
        const int t = rev ? (T_LEN - 1 - step) : step;
        const float* xgt = xg + (size_t)t * BATCH * G4;
        if (step + 1 < T_LEN) {
            const int tn = rev ? (T_LEN - 2 - step) : step + 1;
            __builtin_prefetch(xg + (size_t)tn * BATCH * G4 + tid * 128, 0, 0);
        }
        // ---- g = h @ Whh^T + xg : 16 waves x 16 tiles x (256/32) WMMA ----
        for (int jj = 0; jj < 16; ++jj) {
            const int nt = ntb + 4 * jj;
            FragC acc;
#pragma unroll
            for (int i = 0; i < 8; ++i) acc.f[i] = 0.0f;
            for (int kt = 0; kt < 8; ++kt) {
                FragAB a, bf;
                const unsigned int* ap = hA + ((mt * 8 + kt) * 32 + lane) * 8;
                const unsigned int* bp = whhp + (((size_t)nt * 8 + kt) * 32 + lane) * 8;
#pragma unroll
                for (int v = 0; v < 8; ++v) { a.u[v] = ap[v]; bf.u[v] = bp[v]; }
                acc.v = __builtin_amdgcn_wmma_f32_16x16x32_bf16(false, a.v, false, bf.v,
                                                                (short)0, acc.v, false, false);
            }
            const int col = nt * 16 + l16;
#pragma unroll
            for (int v = 0; v < 8; ++v) {
                int row = mt * 16 + v + 8 * half;
                gbuf[row * G4 + col] = acc.f[v] + xgt[(size_t)row * G4 + col];
            }
        }
        __syncthreads();
        // ---- gates + state update; each thread owns 32 cells ----
        float hv[32];
        const float* gb = gbuf + b * G4;
#pragma unroll
        for (int j = 0; j < 32; ++j) {
            int hd = hd0 + j;
            float ig = sigf(gb[hd]);
            float fg = sigf(gb[HDIM + hd]);
            float gg = tanhf(gb[2 * HDIM + hd]);
            float og = sigf(gb[3 * HDIM + hd]);
            float c = fg * creg[j] + ig * gg;
            creg[j] = c;
            hv[j] = og * tanhf(c);
        }
        float* orow = xout + ((size_t)t * BATCH + b) * 512 + dir * HDIM + hd0;
        unsigned int* brow = xoutbf + ((size_t)t * BATCH + b) * 256 + dir * 128 + kto * 16;
#pragma unroll
        for (int j = 0; j < 32; ++j) orow[j] = hv[j];
#pragma unroll
        for (int j = 0; j < 32; j += 2) {
            brow[j >> 1] = pack2bf(hv[j], hv[j + 1]);    // bf16 copy for next GEMM
            int la = (b & 15) + 16 * ((j >> 3) & 1);
            int v  = ((j & 7) >> 1) + 4 * (j >> 4);
            hA[(((b >> 4) * 8 + kto) * 32 + la) * 8 + v] = pack2bf(hv[j], hv[j + 1]);
        }
        __syncthreads();
    }
}

// ------------- feats = x @ w_out^T + b_out  (K=7, trivial VALU) -------------
__global__ void k_tagproj(const float* __restrict__ x, const float* __restrict__ w,
                          const float* __restrict__ bo, float* __restrict__ feats) {
    int id = blockIdx.x * blockDim.x + threadIdx.x;
    if (id >= T_LEN * BATCH * K_TAGS) return;
    int row = id / K_TAGS, k = id - row * K_TAGS;
    const float4* xr = (const float4*)(x + (size_t)row * 512);
    const float4* wr = (const float4*)(w + (size_t)k * 512);
    float acc = bo[k];
    for (int e = 0; e < 128; ++e) {
        float4 a = xr[e], c = wr[e];
        acc += a.x * c.x + a.y * c.y + a.z * c.z + a.w * c.w;
    }
    feats[id] = acc;
}

// ------------------------- CRF Viterbi, 1 thread per batch -------------------------
__global__ void k_viterbi(const float* __restrict__ feats, const float* __restrict__ trans,
                          unsigned char* __restrict__ bp, float* __restrict__ out) {
    __shared__ float tr[K_TAGS * K_TAGS];
    if (threadIdx.x < K_TAGS * K_TAGS) tr[threadIdx.x] = trans[threadIdx.x];
    __syncthreads();
    const int b = threadIdx.x;
    if (b >= BATCH) return;
    float v[K_TAGS];
#pragma unroll
    for (int k = 0; k < K_TAGS; ++k) v[k] = NEGINF;
    v[START_TAG] = 0.0f;
    for (int t = 0; t < T_LEN; ++t) {
        const float* ft = feats + ((size_t)t * BATCH + b) * K_TAGS;
        float nv[K_TAGS];
#pragma unroll
        for (int nk = 0; nk < K_TAGS; ++nk) {
            float best = v[0] + tr[nk * K_TAGS + 0];
            int arg = 0;
#pragma unroll
            for (int pk = 1; pk < K_TAGS; ++pk) {
                float s = v[pk] + tr[nk * K_TAGS + pk];
                if (s > best) { best = s; arg = pk; }
            }
            bp[((size_t)t * BATCH + b) * K_TAGS + nk] = (unsigned char)arg;
            nv[nk] = best + ft[nk];
        }
#pragma unroll
        for (int k = 0; k < K_TAGS; ++k) v[k] = nv[k];
    }
    float best = v[0] + tr[STOP_TAG * K_TAGS + 0];
    int tag = 0;
#pragma unroll
    for (int k = 1; k < K_TAGS; ++k) {
        float s = v[k] + tr[STOP_TAG * K_TAGS + k];
        if (s > best) { best = s; tag = k; }
    }
    int* path = (int*)out;
    float* score = out + BATCH * T_LEN;
    score[b] = best;
    for (int t = T_LEN - 1; t >= 0; --t) {
        path[(size_t)b * T_LEN + t] = tag;
        tag = bp[((size_t)t * BATCH + b) * K_TAGS + tag];
    }
}

extern "C" void kernel_launch(void* const* d_in, const int* in_sizes, int n_in,
                              void* d_out, int out_size, void* d_ws, size_t ws_size,
                              hipStream_t stream) {
    (void)in_sizes; (void)n_in; (void)out_size; (void)ws_size;
    const int*   sent  = (const int*)  d_in[0];
    const float* emb   = (const float*)d_in[1];
    const float* w_ih  = (const float*)d_in[2];
    const float* w_hh  = (const float*)d_in[3];
    const float* b_ih  = (const float*)d_in[4];
    const float* b_hh  = (const float*)d_in[5];
    const float* w_out = (const float*)d_in[6];
    const float* b_out = (const float*)d_in[7];
    const float* trans = (const float*)d_in[8];
    const float* h0    = (const float*)d_in[9];
    const float* c0    = (const float*)d_in[10];

    char* ws = (char*)d_ws;
    unsigned int* xbfA = (unsigned int*)(ws);                       // 16,777,216 B
    unsigned int* xbfB = (unsigned int*)(ws + (size_t)16777216);    // 16,777,216 B
    float* xg0  = (float*)(ws + (size_t)33554432);                  // 67,108,864 B
    float* xg1  = (float*)(ws + (size_t)100663296);                 // 67,108,864 B
    float* fbuf = (float*)(ws + (size_t)167772160);                 // 33,554,432 B
    unsigned int* whhp  = (unsigned int*)(ws + (size_t)201326592);  //  2,097,152 B
    unsigned int* wihp  = (unsigned int*)(ws + (size_t)203423744);  //  4,194,304 B
    float*         feats = (float*)(ws + (size_t)207618048);        //    458,752 B
    unsigned char* bptr  = (unsigned char*)(ws + (size_t)208076800);//    114,688 B

    k_prep_whh<<<(4 * 131072 + 255) / 256, 256, 0, stream>>>(w_hh, whhp);
    k_prep_wih<<<(4 * 262144 + 255) / 256, 256, 0, stream>>>(w_ih, wihp);
    k_embed<<<T_LEN * BATCH, 128, 0, stream>>>(sent, emb, xbfA);

    const size_t scan_lds = (size_t)BATCH * G4 * 4 + (size_t)4 * 8 * 32 * 8 * 4; // 288 KB
    unsigned int* xin = xbfA; unsigned int* xob = xbfB;
    for (int l = 0; l < 2; ++l) {
        k_gemm_xw<<<dim3(1024, 16, 2), 128, 0, stream>>>(
            xin, wihp + (size_t)l * 2 * 262144,
            b_ih + (size_t)l * 2 * G4, b_hh + (size_t)l * 2 * G4, xg0, xg1);
        k_lstm_scan<<<2, 512, scan_lds, stream>>>(
            xg0, xg1, whhp + (size_t)l * 262144,
            h0 + (size_t)2 * l * BATCH * HDIM, c0 + (size_t)2 * l * BATCH * HDIM,
            fbuf, xob);
        unsigned int* tmp = xin; xin = xob; xob = tmp;
    }
    k_tagproj<<<(T_LEN * BATCH * K_TAGS + 255) / 256, 256, 0, stream>>>(fbuf, w_out, b_out, feats);
    k_viterbi<<<1, 64, 0, stream>>>(feats, trans, bptr, (float*)d_out);
}
